// MultiheadFlashSelfAttention_90400471646735
// MI455X (gfx1250) — matrix operand order repaired and checked by reading, not compiled, not run on hardware
//
#include <hip/hip_runtime.h>
#include <hip/hip_bf16.h>

typedef __bf16 bf16_t;
typedef __attribute__((ext_vector_type(4)))  __bf16 v4bf;
typedef __attribute__((ext_vector_type(8)))  __bf16 v8bf;
typedef __attribute__((ext_vector_type(16))) __bf16 v16bf;
typedef __attribute__((ext_vector_type(8)))  float  v8f;

__device__ __forceinline__ v16bf combine16(v8bf lo, v8bf hi) {
    union { v16bf v; v8bf h[2]; } u;
    u.h[0] = lo; u.h[1] = hi;
    return u.v;
}
__device__ __forceinline__ v8bf ldg8(const bf16_t* p) { return *(const v8bf*)p; }

// Low 32 bits of a generic pointer to __shared__ = wave-relative LDS offset
// (flat LDS aperture encodes the offset in addr[31:0], ISA 10.2).
__device__ __forceinline__ unsigned lds_addr32(const void* p) {
    return (unsigned)(unsigned long long)p;
}

// Async copy 16B global -> LDS (ASYNCcnt-tracked, no VGPR data).
__device__ __forceinline__ void async_g2l_b128(const void* gp, const void* lp) {
    asm volatile("global_load_async_to_lds_b128 %0, %1, off"
                 :: "v"(lds_addr32(lp)), "v"((unsigned long long)gp)
                 : "memory");
}

// 16-lane butterfly reductions on the VALU via v_permlane16_b32
// (sel[i] = i ^ mask, packed as nibbles; lanes 0-15 / 16-31 reduce separately).
__device__ __forceinline__ float bfly16(float v, unsigned lo, unsigned hi) {
    unsigned s = __builtin_bit_cast(unsigned, v);
    unsigned r = __builtin_amdgcn_permlane16(s, s, lo, hi, false, false);
    return __builtin_bit_cast(float, r);
}
__device__ __forceinline__ float rowmax16(float v) {
    v = fmaxf(v, bfly16(v, 0x67452301u, 0xEFCDAB89u));   // xor 1
    v = fmaxf(v, bfly16(v, 0x54761032u, 0xDCFE98BAu));   // xor 2
    v = fmaxf(v, bfly16(v, 0x32107654u, 0xBA98FEDCu));   // xor 4
    v = fmaxf(v, bfly16(v, 0xFEDCBA98u, 0x76543210u));   // xor 8
    return v;
}
__device__ __forceinline__ float rowsum16(float v) {
    v += bfly16(v, 0x67452301u, 0xEFCDAB89u);
    v += bfly16(v, 0x54761032u, 0xDCFE98BAu);
    v += bfly16(v, 0x32107654u, 0xBA98FEDCu);
    v += bfly16(v, 0xFEDCBA98u, 0x76543210u);
    return v;
}

// ---------------------------------------------------------------------------
// f32 -> bf16 cast, 4 elements / thread
// ---------------------------------------------------------------------------
__global__ __launch_bounds__(256) void cast_f32_bf16_kernel(
    const float* __restrict__ in, bf16_t* __restrict__ out, int n)
{
    int i = (blockIdx.x * blockDim.x + threadIdx.x) * 4;
    if (i < n) {
        float4 v = *(const float4*)(in + i);
        v4bf o;
        o[0] = (bf16_t)v.x; o[1] = (bf16_t)v.y;
        o[2] = (bf16_t)v.z; o[3] = (bf16_t)v.w;
        *(v4bf*)(out + i) = o;
    }
}

// ---------------------------------------------------------------------------
// GEMM  C[m][n] = sum_k A[m][k] * W[n][k]   (A: MxK bf16, W: NxK bf16)
// 128x64 workgroup tile, 8 waves x (16x64), double-buffered LDS tiles filled
// by GLOBAL_LOAD_ASYNC_TO_LDS_B128 so tile i+1 streams while tile i computes.
// Each wave issues 3 async loads per tile; loads complete in order, so
// s_wait_asynccnt 3 after issuing the next tile means the current one landed.
// ---------------------------------------------------------------------------
template<bool OUT_BF16>
__global__ __launch_bounds__(256) void gemm_nt_kernel(
    const bf16_t* __restrict__ A, const bf16_t* __restrict__ W,
    void* __restrict__ Cout, int M, int N, int K)
{
    __shared__ bf16_t As[2][128][40];   // 32 + 8 pad
    __shared__ bf16_t Ws[2][64][40];

    const int tid  = threadIdx.x;
    const int wave = tid >> 5, lane = tid & 31;
    const int lm = lane & 15, lh = lane >> 4;
    const int m0 = blockIdx.y * 128;
    const int n0 = blockIdx.x * 64;

    v8f acc[4];
    #pragma unroll
    for (int t = 0; t < 4; ++t)
        #pragma unroll
        for (int i = 0; i < 8; ++i) acc[t][i] = 0.f;

    const int arow = tid >> 1;          // 0..127
    const int acol = (tid & 1) * 16;    // 0 or 16
    const int wrow = tid >> 2;          // 0..63
    const int wcol = (tid & 3) * 8;     // 0..24

    auto stage = [&](int buf, int kb) {
        const bf16_t* ag = A + (size_t)(m0 + arow) * K + kb + acol;
        async_g2l_b128(ag,     &As[buf][arow][acol]);
        async_g2l_b128(ag + 8, &As[buf][arow][acol + 8]);
        const bf16_t* wg = W + (size_t)(n0 + wrow) * K + kb + wcol;
        async_g2l_b128(wg, &Ws[buf][wrow][wcol]);
    };

    stage(0, 0);
    for (int kb = 0; kb < K; kb += 32) {
        const int buf = (kb >> 5) & 1;
        if (kb + 32 < K) {
            stage(buf ^ 1, kb + 32);
            asm volatile("s_wait_asynccnt 3" ::: "memory");   // current tile done
        } else {
            asm volatile("s_wait_asynccnt 0" ::: "memory");
        }
        __syncthreads();

        v16bf afrag = combine16(*(const v8bf*)&As[buf][wave * 16 + lm][8 * lh],
                                *(const v8bf*)&As[buf][wave * 16 + lm][8 * lh + 16]);
        #pragma unroll
        for (int t = 0; t < 4; ++t) {
            v16bf bfrag = combine16(*(const v8bf*)&Ws[buf][t * 16 + lm][8 * lh],
                                    *(const v8bf*)&Ws[buf][t * 16 + lm][8 * lh + 16]);
            acc[t] = __builtin_amdgcn_wmma_f32_16x16x32_bf16(
                false, afrag, false, bfrag, (short)0, acc[t], false, false);
        }
        __syncthreads();    // protect buf from being overwritten by next stage
    }

    #pragma unroll
    for (int t = 0; t < 4; ++t) {
        #pragma unroll
        for (int r = 0; r < 8; ++r) {
            int row = m0 + wave * 16 + r + 8 * lh;
            int col = n0 + t * 16 + lm;
            float v = acc[t][r];
            if (OUT_BF16) ((bf16_t*)Cout)[(size_t)row * N + col] = (bf16_t)v;
            else          ((float*)Cout)[(size_t)row * N + col]  = v;
        }
    }
}

// ---------------------------------------------------------------------------
// RoPE on Q and K in place (bf16 storage, f32 math), interleaved pairs.
// The attention scale 1/sqrt(dk) = 0.125 is folded into Q here.
// ---------------------------------------------------------------------------
__global__ __launch_bounds__(256) void rope_kernel(
    bf16_t* __restrict__ Q, bf16_t* __restrict__ Km,
    const int* __restrict__ pos, int S, int D, int total)
{
    int p = blockIdx.x * blockDim.x + threadIdx.x;
    if (p >= total) return;
    int m = p >> 9;          // row in [0, B*S)
    int c = p & 511;
    int h = c >> 5;          // head
    int i = c & 31;          // pair index (half = 32)
    int s = m & (S - 1);     // S = 2048
    float freq = __powf(10000.f, -(float)i * (1.f / 32.f));
    float ang = (float)pos[s] * freq;
    float sn, cs;
    __sincosf(ang, &sn, &cs);
    size_t base = (size_t)m * D + h * 64 + 2 * i;

    const float qscale = 0.125f;        // 1/sqrt(64)
    float q1 = (float)Q[base], q2 = (float)Q[base + 1];
    Q[base]     = (bf16_t)((q1 * cs - q2 * sn) * qscale);
    Q[base + 1] = (bf16_t)((q1 * sn + q2 * cs) * qscale);
    float k1 = (float)Km[base], k2 = (float)Km[base + 1];
    Km[base]     = (bf16_t)(k1 * cs - k2 * sn);
    Km[base + 1] = (bf16_t)(k1 * sn + k2 * cs);
}

// ---------------------------------------------------------------------------
// Online-softmax update for one 16x32 score block.
// MASKED=true for a wave's diagonal j-block and the fully masked blocks after it;
// all strictly-lower blocks take the maskless fast path (no v_cmp/v_cndmask).
// ---------------------------------------------------------------------------
template<bool MASKED>
__device__ __forceinline__ void softmax_step(
    const v8f (&sacc)[2], float (&mrow)[8], float (&lrow)[8], v8f (&oacc)[4],
    bf16_t (*PsW)[40], int qrow, int j0, int lm, int lh)
{
    const int jc0 = j0 + lm, jc1 = j0 + 16 + lm;
    #pragma unroll
    for (int r = 0; r < 8; ++r) {
        const int q = qrow + r + 8 * lh;
        bool k0 = true, k1 = true;
        float s0 = sacc[0][r], s1 = sacc[1][r];
        if (MASKED) {
            k0 = jc0 <= q; k1 = jc1 <= q;
            s0 = k0 ? s0 : -3.0e38f;
            s1 = k1 ? s1 : -3.0e38f;
        }
        float mt = rowmax16(fmaxf(s0, s1));
        float mnew = fmaxf(mrow[r], mt);
        float p0 = __expf(s0 - mnew);
        float p1 = __expf(s1 - mnew);
        if (MASKED) { p0 = k0 ? p0 : 0.f; p1 = k1 ? p1 : 0.f; }
        float corr = __expf(mrow[r] - mnew);
        float psum = rowsum16(p0 + p1);
        lrow[r] = lrow[r] * corr + psum;
        mrow[r] = mnew;
        #pragma unroll
        for (int t = 0; t < 4; ++t) oacc[t][r] *= corr;
        PsW[r + 8 * lh][lm]      = (bf16_t)p0;
        PsW[r + 8 * lh][16 + lm] = (bf16_t)p1;
    }
}

// ---------------------------------------------------------------------------
// Causal flash attention for one (b,h): 8 waves x 16 q-rows = 128 rows/block.
// K block staged by async-to-LDS; V block staged transposed (register ->
// ds_store_b16) so the PV B-fragment read is a contiguous ds_load_b128.
// P goes through a per-wave LDS tile (C-layout -> A-layout) guarded by an
// explicit s_wait_dscnt for the same-wave store->load hazard.
// Row reductions are v_permlane16 butterflies (VALU, co-executes with WMMA).
// ---------------------------------------------------------------------------
__global__ __launch_bounds__(256) void flash_attn_kernel(
    const bf16_t* __restrict__ Q, const bf16_t* __restrict__ Km,
    const bf16_t* __restrict__ V, bf16_t* __restrict__ O, int S, int D)
{
    __shared__ bf16_t Ks[32][72];       // [j][d], pad 8
    __shared__ bf16_t Vt[64][40];       // [d][j], pad 8
    __shared__ bf16_t Ps[8][16][40];    // per-wave P tile [q][j], pad 8

    const int tid  = threadIdx.x;
    const int wave = tid >> 5, lane = tid & 31;
    const int lm = lane & 15, lh = lane >> 4;
    const int bh = blockIdx.y;          // b*16 + h
    const int b = bh >> 4, h = bh & 15;
    const int q0 = blockIdx.x * 128;
    const int qrow = q0 + wave * 16;

    // Q fragments (Q pre-scaled by 1/sqrt(dk) in rope_kernel)
    const size_t rowQ = ((size_t)(b * S + qrow + lm)) * D + h * 64;
    v16bf aQ[2];
    #pragma unroll
    for (int ks = 0; ks < 2; ++ks)
        aQ[ks] = combine16(ldg8(Q + rowQ + 32 * ks + 8 * lh),
                           ldg8(Q + rowQ + 32 * ks + 8 * lh + 16));

    float mrow[8], lrow[8];
    v8f oacc[4];
    #pragma unroll
    for (int r = 0; r < 8; ++r) { mrow[r] = -3.0e38f; lrow[r] = 0.f; }
    #pragma unroll
    for (int t = 0; t < 4; ++t)
        #pragma unroll
        for (int i = 0; i < 8; ++i) oacc[t][i] = 0.f;

    const int jend = q0 + 128;          // causal bound, uniform per block
    const int krow = tid >> 3;          // 0..31
    const int kcol = (tid & 7) * 8;     // 0..56

    for (int j0 = 0; j0 < jend; j0 += 32) {
        const size_t src = ((size_t)(b * S + j0 + krow)) * D + h * 64 + kcol;
        // K block via async copy (no VGPR round trip)
        async_g2l_b128(Km + src, &Ks[krow][kcol]);
        // V block transposed through registers
        v8bf vv = ldg8(V + src);
        #pragma unroll
        for (int i = 0; i < 8; ++i) Vt[kcol + i][krow] = vv[i];
        // prefetch next j-block while we compute this one
        if (j0 + 32 < jend) {
            __builtin_prefetch(Km + src + (size_t)32 * D, 0, 1);
            __builtin_prefetch(V  + src + (size_t)32 * D, 0, 1);
        }
        asm volatile("s_wait_asynccnt 0" ::: "memory");
        __syncthreads();

        // S = Q K^T : two 16x16 score tiles over j, two K=32 steps over d
        v8f sacc[2];
        #pragma unroll
        for (int t = 0; t < 2; ++t)
            #pragma unroll
            for (int i = 0; i < 8; ++i) sacc[t][i] = 0.f;
        #pragma unroll
        for (int ks = 0; ks < 2; ++ks) {
            #pragma unroll
            for (int t = 0; t < 2; ++t) {
                v16bf bfrag = combine16(
                    *(const v8bf*)&Ks[t * 16 + lm][32 * ks + 8 * lh],
                    *(const v8bf*)&Ks[t * 16 + lm][32 * ks + 8 * lh + 16]);
                sacc[t] = __builtin_amdgcn_wmma_f32_16x16x32_bf16(
                    false, aQ[ks], false, bfrag, (short)0, sacc[t], false, false);
            }
        }

        // online softmax; wave-uniform branch keeps EXEC all-1s around WMMA.
        // Fully-unmasked block iff worst element (q=qrow, j=j0+31) is legal.
        if (j0 + 31 <= qrow)
            softmax_step<false>(sacc, mrow, lrow, oacc, Ps[wave], qrow, j0, lm, lh);
        else
            softmax_step<true >(sacc, mrow, lrow, oacc, Ps[wave], qrow, j0, lm, lh);

        // same-wave LDS store -> load hazard (per-wave tile, no barrier)
        asm volatile("s_wait_dscnt 0" ::: "memory");

        // O += P V : P as A-fragment (16x32); V^T rows give contiguous B-frag
        v16bf pfrag = combine16(*(const v8bf*)&Ps[wave][lm][8 * lh],
                                *(const v8bf*)&Ps[wave][lm][8 * lh + 16]);
        #pragma unroll
        for (int t = 0; t < 4; ++t) {
            v16bf bfrag = combine16(*(const v8bf*)&Vt[t * 16 + lm][8 * lh],
                                    *(const v8bf*)&Vt[t * 16 + lm][8 * lh + 16]);
            oacc[t] = __builtin_amdgcn_wmma_f32_16x16x32_bf16(
                false, pfrag, false, bfrag, (short)0, oacc[t], false, false);
        }
        __syncthreads();
    }

    // normalize and store attention output (bf16, feeds Wo GEMM)
    #pragma unroll
    for (int t = 0; t < 4; ++t) {
        #pragma unroll
        for (int r = 0; r < 8; ++r) {
            int q = qrow + r + 8 * lh;
            float v = oacc[t][r] / lrow[r];
            O[((size_t)(b * S + q)) * D + h * 64 + t * 16 + lm] = (bf16_t)v;
        }
    }
}

// ---------------------------------------------------------------------------
extern "C" void kernel_launch(void* const* d_in, const int* in_sizes, int n_in,
                              void* d_out, int out_size, void* d_ws, size_t ws_size,
                              hipStream_t stream)
{
    const float* x   = (const float*)d_in[0];
    const int* tpos  = (const int*)d_in[1];
    const float* Wq  = (const float*)d_in[2];
    const float* Wk  = (const float*)d_in[3];
    const float* Wv  = (const float*)d_in[4];
    const float* Wo  = (const float*)d_in[5];

    const int B = 4, S = 2048, D = 1024;
    const int M = B * S;                    // 8192

    // workspace carve-up (bf16 buffers)
    char* w = (char*)d_ws;
    bf16_t* xb  = (bf16_t*)w; w += (size_t)M * D * 2;
    bf16_t* wqb = (bf16_t*)w; w += (size_t)D * D * 2;
    bf16_t* wkb = (bf16_t*)w; w += (size_t)D * D * 2;
    bf16_t* wvb = (bf16_t*)w; w += (size_t)D * D * 2;
    bf16_t* wob = (bf16_t*)w; w += (size_t)D * D * 2;
    bf16_t* qb  = (bf16_t*)w; w += (size_t)M * D * 2;
    bf16_t* kb  = (bf16_t*)w; w += (size_t)M * D * 2;
    bf16_t* vb  = (bf16_t*)w; w += (size_t)M * D * 2;
    bf16_t* ab  = (bf16_t*)w; w += (size_t)M * D * 2;

    // 1. cast inputs to bf16
    {
        int nx = M * D;
        cast_f32_bf16_kernel<<<(nx / 4 + 255) / 256, 256, 0, stream>>>(x, xb, nx);
        int nw = D * D;
        int blocks = (nw / 4 + 255) / 256;
        cast_f32_bf16_kernel<<<blocks, 256, 0, stream>>>(Wq, wqb, nw);
        cast_f32_bf16_kernel<<<blocks, 256, 0, stream>>>(Wk, wkb, nw);
        cast_f32_bf16_kernel<<<blocks, 256, 0, stream>>>(Wv, wvb, nw);
        cast_f32_bf16_kernel<<<blocks, 256, 0, stream>>>(Wo, wob, nw);
    }

    // 2. Q/K/V projections (bf16 out)
    dim3 gg(D / 64, M / 128);               // (16, 64)
    gemm_nt_kernel<true><<<gg, 256, 0, stream>>>(xb, wqb, qb, M, D, D);
    gemm_nt_kernel<true><<<gg, 256, 0, stream>>>(xb, wkb, kb, M, D, D);
    gemm_nt_kernel<true><<<gg, 256, 0, stream>>>(xb, wvb, vb, M, D, D);

    // 3. RoPE on Q, K (Q additionally pre-scaled by 1/sqrt(dk))
    {
        int nr = M * 512;                   // rows * H * 32 pairs
        rope_kernel<<<(nr + 255) / 256, 256, 0, stream>>>(qb, kb, tpos, S, D, nr);
    }

    // 4. causal flash attention
    dim3 ga(S / 128, B * 16);               // (16, 64)
    flash_attn_kernel<<<ga, 256, 0, stream>>>(qb, kb, vb, ab, S, D);

    // 5. output projection (f32 out to d_out)
    gemm_nt_kernel<false><<<gg, 256, 0, stream>>>(ab, wob, (float*)d_out, M, D, D);
}
